// BahdanauAttention_75067438400088
// MI455X (gfx1250) — matrix operand order repaired and checked by reading, not compiled, not run on hardware
//
#include <hip/hip_runtime.h>
#include <hip/hip_bf16.h>

// Problem constants (match reference)
#define BB 32
#define SS 2048
#define HH 1024
#define KT 32      // K per WMMA step
#define ST 32      // s-columns per workgroup

typedef __attribute__((ext_vector_type(16))) __bf16 v16bf;
typedef __attribute__((ext_vector_type(8)))  float  v8f;

__device__ __forceinline__ unsigned short f2bf(float f) {
    unsigned int u = __float_as_uint(f);
    unsigned int r = u + 0x7fffu + ((u >> 16) & 1u);   // round-to-nearest-even
    return (unsigned short)(r >> 16);
}

// ---------------- Kernel 1: W1 fp32 -> bf16 (L2-resident operand) ----------------
__global__ __launch_bounds__(256)
void cvt_w1_kernel(const float* __restrict__ in, unsigned short* __restrict__ out) {
    int i = (blockIdx.x * 256 + threadIdx.x) * 4;          // H*H elements, 4 per thread
    float4 f = *reinterpret_cast<const float4*>(in + i);
    unsigned int lo = (unsigned int)f2bf(f.x) | ((unsigned int)f2bf(f.y) << 16);
    unsigned int hi = (unsigned int)f2bf(f.z) | ((unsigned int)f2bf(f.w) << 16);
    *reinterpret_cast<uint2*>(out + i) = make_uint2(lo, hi);
}

// ---------------- Kernel 2: W2h[b,o] = sum_h hidden[b,h] * W2[o,h] ----------------
__global__ __launch_bounds__(256)
void w2h_kernel(const float* __restrict__ hidden, const float* __restrict__ W2,
                float* __restrict__ out) {
    int o = blockIdx.x * 256 + threadIdx.x;
    int b = blockIdx.y;
    const float* hrow = hidden + (size_t)b * HH;
    const float* wrow = W2 + (size_t)o * HH;
    float acc = 0.0f;
    for (int h = 0; h < HH; h += 4) {
        float4 w = *reinterpret_cast<const float4*>(wrow + h);
        float4 x = *reinterpret_cast<const float4*>(hrow + h);
        acc = fmaf(w.x, x.x, fmaf(w.y, x.y, fmaf(w.z, x.z, fmaf(w.w, x.w, acc))));
    }
    out[(size_t)b * HH + o] = acc;
}

// ---------------- Kernel 3: fused score = v . tanh(W1 @ e + W2h), via WMMA bf16 ----
// Grid: (SS/ST, BB), block 512 (16 waves). Each WG: 32 s-cols x full 1024 o, K outer.
// LDS: lA[1024][32] bf16 (64KB) + lB[32][32] bf16 (2KB) + lpart[16][32] f32 (2KB).
#define SMEM_A   0
#define SMEM_B   (1024 * 32 * 2)
#define SMEM_P   (SMEM_B + 32 * 32 * 2)
#define SMEM_TOT (SMEM_P + 16 * 32 * 4)

__global__ __launch_bounds__(512)
void score_kernel(const float* __restrict__ enc,            // [B,S,H]
                  const unsigned short* __restrict__ W1bf,  // [H,H] bf16
                  const float* __restrict__ w2h,            // [B,H]
                  const float* __restrict__ vvec,           // [H]
                  float* __restrict__ scores) {             // [B,S] raw scores
    extern __shared__ unsigned char smem[];
    unsigned short* lA = reinterpret_cast<unsigned short*>(smem + SMEM_A);
    unsigned short* lB = reinterpret_cast<unsigned short*>(smem + SMEM_B);
    float*       lpart = reinterpret_cast<float*>(smem + SMEM_P);

    const int t    = threadIdx.x;
    const int b    = blockIdx.y;
    const int s0   = blockIdx.x * ST;
    const int wave = t >> 5;
    const int lane = t & 31;
    const int g    = lane >> 4;    // lane half (A: K-group, B: K-group, C: M-group)
    const int nl   = lane & 15;    // A: M row, B/C: N column
    const int sw   = wave & 1;     // which 16-col s-subtile this wave owns
    const int ow   = wave >> 1;    // o-subtile (0..7) within each 128-row block

    v8f acc[8];
#pragma unroll
    for (int j = 0; j < 8; ++j) acc[j] = (v8f){0, 0, 0, 0, 0, 0, 0, 0};

    const float* encB = enc + ((size_t)b * SS + s0) * HH;

    // Per-thread staging addresses
    const int sr = t >> 4;                    // encoder row (0..31), 2 floats/thread
    const int kc = (t & 15) * 2;              // encoder col pair
    const int ar = t >> 2;                    // W1 base row (0..127), +i*128
    const int aq = (t & 3) * 8;               // W1 k-chunk (8 bf16 = 16B)

    for (int k0 = 0; k0 < HH; k0 += KT) {
        __syncthreads();
        // Stage encoder tile [32s x 32k], fp32 -> bf16 on the fly
        {
            const float* p = encB + (size_t)sr * HH + k0 + kc;
            float2 f = *reinterpret_cast<const float2*>(p);
            __builtin_prefetch(p + KT, 0, 0);   // next K tile -> global_prefetch_b8
            lB[sr * 32 + kc]     = f2bf(f.x);
            lB[sr * 32 + kc + 1] = f2bf(f.y);
        }
        // Stage W1 slab [1024o x 32k] bf16 (64KB), 8 x 16B chunks per thread
        {
            const uint4* src =
                reinterpret_cast<const uint4*>(W1bf + (size_t)ar * HH + k0 + aq);
            uint4* dst = reinterpret_cast<uint4*>(&lA[ar * 32 + aq]);
#pragma unroll
            for (int i = 0; i < 8; ++i)
                dst[i * (128 * 32 / 8)] = src[(size_t)i * (128 * HH / 8)];
        }
        __syncthreads();

        // B fragment (K=32 x N=16): lane holds 16 contiguous K at g*16 for col nl
        union { uint4 q[2]; v16bf v; } bf;
        {
            const uint4* pb =
                reinterpret_cast<const uint4*>(&lB[(sw * 16 + nl) * 32 + g * 16]);
            bf.q[0] = pb[0];
            bf.q[1] = pb[1];
        }
#pragma unroll
        for (int j = 0; j < 8; ++j) {
            // A fragment (M=16 x K=32): row = o, 16 contiguous K at g*16
            union { uint4 q[2]; v16bf v; } af;
            const unsigned short* pr = &lA[(j * 128 + ow * 16 + nl) * 32];
            af.q[0] = *reinterpret_cast<const uint4*>(pr + g * 16);
            af.q[1] = *reinterpret_cast<const uint4*>(pr + g * 16 + 8);
            acc[j] = __builtin_amdgcn_wmma_f32_16x16x32_bf16(
                false, af.v, false, bf.v, (short)0, acc[j], false, false);
        }
    }

    // Epilogue: partial score for column (s0 + sw*16 + nl) from this lane's 64 o rows
    float partial = 0.0f;
    const float* w2hb = w2h + (size_t)b * HH;
#pragma unroll
    for (int j = 0; j < 8; ++j) {
        int ob = j * 128 + ow * 16 + g * 8;   // 8 consecutive o rows (C: M = elem + g*8)
        alignas(16) float wv[8], vv[8];
        *reinterpret_cast<float4*>(&wv[0]) = *reinterpret_cast<const float4*>(w2hb + ob);
        *reinterpret_cast<float4*>(&wv[4]) = *reinterpret_cast<const float4*>(w2hb + ob + 4);
        *reinterpret_cast<float4*>(&vv[0]) = *reinterpret_cast<const float4*>(vvec + ob);
        *reinterpret_cast<float4*>(&vv[4]) = *reinterpret_cast<const float4*>(vvec + ob + 4);
#pragma unroll
        for (int i = 0; i < 8; ++i)
            partial += tanhf(acc[j][i] + wv[i]) * vv[i];
    }

    // Deterministic cross-wave reduction (fixed summation order, no FP atomics)
    lpart[wave * 32 + lane] = partial;
    __syncthreads();
    if (t < 32) {
        int swc = t >> 4, nlc = t & 15;
        float sum = 0.0f;
#pragma unroll
        for (int w = 0; w < 8; ++w) {
            int wv = w * 2 + swc;
            sum += lpart[wv * 32 + nlc] + lpart[wv * 32 + nlc + 16];
        }
        scores[(size_t)b * SS + s0 + t] = sum;
    }
}

// ---------------- Kernel 4: masked softmax over S, in place on attn buffer --------
__global__ __launch_bounds__(256)
void softmax_kernel(const unsigned char* __restrict__ mask, float* __restrict__ attn) {
    __shared__ float red[256];
    int b = blockIdx.x, t = threadIdx.x;
    float vals[8];
    float mx = -3.4e38f;
#pragma unroll
    for (int i = 0; i < 8; ++i) {
        int s = t + i * 256;
        float sc = attn[(size_t)b * SS + s];
        sc = mask[(size_t)b * SS + s] ? sc : -1e9f;
        vals[i] = sc;
        mx = fmaxf(mx, sc);
    }
    red[t] = mx; __syncthreads();
    for (int off = 128; off; off >>= 1) {
        if (t < off) red[t] = fmaxf(red[t], red[t + off]);
        __syncthreads();
    }
    mx = red[0]; __syncthreads();
    float sum = 0.0f;
#pragma unroll
    for (int i = 0; i < 8; ++i) { vals[i] = __expf(vals[i] - mx); sum += vals[i]; }
    red[t] = sum; __syncthreads();
    for (int off = 128; off; off >>= 1) {
        if (t < off) red[t] = red[t] + red[t + off];
        __syncthreads();
    }
    float inv = 1.0f / red[0];
#pragma unroll
    for (int i = 0; i < 8; ++i) attn[(size_t)b * SS + t + i * 256] = vals[i] * inv;
}

// ---------------- Kernel 5: context[b,h] = sum_s attn[b,s] * enc[b,s,h] -----------
__global__ __launch_bounds__(256)
void ctx_kernel(const float* __restrict__ enc, const float* __restrict__ attn,
                float* __restrict__ ctx) {
    int h = blockIdx.x * 256 + threadIdx.x;
    int b = blockIdx.y;
    const float* e = enc + (size_t)b * SS * HH + h;
    const float* a = attn + (size_t)b * SS;
    float acc = 0.0f;
    for (int s = 0; s < SS; s += 4) {
        acc = fmaf(a[s + 0], e[(size_t)(s + 0) * HH], acc);
        acc = fmaf(a[s + 1], e[(size_t)(s + 1) * HH], acc);
        acc = fmaf(a[s + 2], e[(size_t)(s + 2) * HH], acc);
        acc = fmaf(a[s + 3], e[(size_t)(s + 3) * HH], acc);
    }
    ctx[(size_t)b * HH + h] = acc;
}

extern "C" void kernel_launch(void* const* d_in, const int* in_sizes, int n_in,
                              void* d_out, int out_size, void* d_ws, size_t ws_size,
                              hipStream_t stream) {
    const float*         hidden = (const float*)d_in[0];          // [B,H]
    const float*         enc    = (const float*)d_in[1];          // [B,S,H]
    const unsigned char* mask   = (const unsigned char*)d_in[2];  // [B,S] bool
    const float*         W1     = (const float*)d_in[3];          // [H,H]
    const float*         W2     = (const float*)d_in[4];          // [H,H]
    const float*         v      = (const float*)d_in[5];          // [H]

    float* ctx  = (float*)d_out;                  // [B,H]
    float* attn = (float*)d_out + BB * HH;        // [B,S] (scores, then softmax)

    // Workspace: W1 in bf16 (2MB) + W2h fp32 (128KB)
    unsigned short* W1bf = (unsigned short*)d_ws;
    float*          w2h  = (float*)((char*)d_ws + (size_t)HH * HH * sizeof(unsigned short));

    cvt_w1_kernel<<<(HH * HH) / (256 * 4), 256, 0, stream>>>(W1, W1bf);
    w2h_kernel<<<dim3(HH / 256, BB), 256, 0, stream>>>(hidden, W2, w2h);
    score_kernel<<<dim3(SS / ST, BB), 512, SMEM_TOT, stream>>>(enc, W1bf, w2h, v, attn);
    softmax_kernel<<<BB, 256, 0, stream>>>(mask, attn);
    ctx_kernel<<<dim3(HH / 256, BB), 256, 0, stream>>>(enc, attn, ctx);
}
